// RNN_59777354825772
// MI455X (gfx1250) — matrix operand order repaired and checked by reading, not compiled, not run on hardware
//
#include <hip/hip_runtime.h>

// ---------------- CDNA5 WMMA types ----------------
typedef __attribute__((ext_vector_type(16))) __bf16 v16bf;
typedef __attribute__((ext_vector_type(8)))  float  v8f;

#define SEQ_LEN 1024
#define BATCH   64
#define INPUT   256
#define HIDDEN  256
#define OUTPUT  64
#define KCH     8          // 256 / 32 K-chunks per WMMA-K sweep

// ---- helpers --------------------------------------------------------------

// f32 -> bf16 via native cast (backend uses v_cvt_*bf16* ops, RNE).
__device__ __forceinline__ unsigned short f2bf_bits(float f) {
    union { __bf16 b; unsigned short s; } u;
    u.b = (__bf16)f;
    return u.s;
}

__device__ __forceinline__ float fast_tanh(float x) {
#if __has_builtin(__builtin_amdgcn_tanhf)
    return __builtin_amdgcn_tanhf(x);        // v_tanh_f32 (TRANS32)
#elif __has_builtin(__builtin_amdgcn_tanh_f32)
    return __builtin_amdgcn_tanh_f32(x);
#else
    return tanhf(x);
#endif
}

// Packed B-operand fragment: 16 bf16 contiguous per lane (32 B).
__device__ __forceinline__ v16bf load_bf16x16(const unsigned short* p) {
    union { v16bf v; uint4 q[2]; } u;
    u.q[0] = *(const uint4*)(p);
    u.q[1] = *(const uint4*)(p + 8);
    return u.v;
}

// A-operand from a 256-wide bf16 row (LDS or global):
// elems 0..7 = K kb+0..7, elems 8..15 = K 16+kb+0..7
__device__ __forceinline__ v16bf load_A_row(const unsigned short* p) {
    union { v16bf v; uint4 q[2]; } u;
    u.q[0] = *(const uint4*)(p);        // +0  bytes
    u.q[1] = *(const uint4*)(p + 16);   // +32 bytes
    return u.v;
}

// ---- kernel 0: prepack weights into WMMA B-operand lane layout ------------
// packed[((nt*KCH + kc)*32 + lane)*16 + e] = bf16(W[n][k]),
//   n = nt*16 + (lane&15),  k = kc*32 + (lane>=16 ? 8 : 0) + e + (e>=8 ? 8 : 0)
__global__ void rnn_prepack(const float* __restrict__ W_in,
                            const float* __restrict__ W_out,
                            unsigned short* __restrict__ Wxp,
                            unsigned short* __restrict__ Whp,
                            unsigned short* __restrict__ Wop) {
    const int NWX = 256 * 256;
    const int NWO = 64 * 256;
    int t = blockIdx.x * blockDim.x + threadIdx.x;
    if (t >= 2 * NWX + NWO) return;

    int arr, idx;
    if (t < NWX)          { arr = 0; idx = t; }
    else if (t < 2 * NWX) { arr = 1; idx = t - NWX; }
    else                  { arr = 2; idx = t - 2 * NWX; }

    int e    = idx & 15;
    int lane = (idx >> 4) & 31;
    int kc   = (idx >> 9) & 7;
    int nt   = idx >> 12;
    int n    = nt * 16 + (lane & 15);
    int k    = kc * 32 + ((lane & 16) ? 8 : 0) + e + ((e & 8) ? 8 : 0);

    float v;
    unsigned short* dst;
    if (arr == 0)      { v = W_in[n * 512 + k];        dst = Wxp + idx; }
    else if (arr == 1) { v = W_in[n * 512 + 256 + k];  dst = Whp + idx; }
    else               { v = W_out[n * 256 + k];       dst = Wop + idx; }
    *dst = f2bf_bits(v);
}

// ---- kernel 0b: convert x (f32) to bf16 once, row-major ------------------
// 8 elems/thread: two b128 loads -> one b128 store of packed bf16.
__global__ void rnn_x2bf(const float* __restrict__ x,
                         unsigned short* __restrict__ xb) {
    size_t i = ((size_t)blockIdx.x * blockDim.x + threadIdx.x) * 8;
    float4 f0 = *(const float4*)(x + i);
    float4 f1 = *(const float4*)(x + i + 4);
    union { uint4 q; unsigned short s[8]; } o;
    o.s[0] = f2bf_bits(f0.x); o.s[1] = f2bf_bits(f0.y);
    o.s[2] = f2bf_bits(f0.z); o.s[3] = f2bf_bits(f0.w);
    o.s[4] = f2bf_bits(f1.x); o.s[5] = f2bf_bits(f1.y);
    o.s[6] = f2bf_bits(f1.z); o.s[7] = f2bf_bits(f1.w);
    *(uint4*)(xb + i) = o.q;
}

// ---- kernel 1: Xproj[s*B+b, :] = x_row @ Wx^T + b_in  (parallel GEMM) -----
// M = 65536 rows, N = 256, K = 256. One 16x16 tile per wave, 8 waves/block.
__global__ void rnn_xproj(const unsigned short* __restrict__ xb,
                          const float* __restrict__ b_in,
                          const unsigned short* __restrict__ Wxp,
                          float* __restrict__ Xproj) {
    const int lane  = threadIdx.x & 31;
    const int wave  = threadIdx.x >> 5;
    const int mt    = blockIdx.x >> 1;                 // 0..4095
    const int nt    = ((blockIdx.x & 1) << 3) + wave;  // 0..15
    const int row_l = lane & 15;
    const int hi    = (lane >> 4) & 1;
    const int kb    = hi ? 8 : 0;

    const size_t m = (size_t)mt * 16 + row_l;          // A row for this lane
    const int    n = nt * 16 + row_l;                  // C column for this lane

    const float bias = b_in[n];
    v8f c;
#pragma unroll
    for (int v = 0; v < 8; ++v) c[v] = bias;

#pragma unroll
    for (int kc = 0; kc < KCH; ++kc) {
        v16bf a = load_A_row(xb + m * 256 + kc * 32 + kb);
        v16bf b = load_bf16x16(Wxp + ((size_t)((nt * KCH + kc) * 32 + lane) << 4));
        c = __builtin_amdgcn_wmma_f32_16x16x32_bf16(false, a, false, b,
                                                    (short)0, c, false, false);
    }
#pragma unroll
    for (int v = 0; v < 8; ++v)
        Xproj[((size_t)mt * 16 + v + hi * 8) * 256 + n] = c[v];
}

// ---- kernel 2: the scan. 4 blocks x 512 threads; block = 16 batch rows ----
__global__ void rnn_scan(const float* __restrict__ pre_state,
                         const float* __restrict__ b_out,
                         const unsigned short* __restrict__ Whp,
                         const unsigned short* __restrict__ Wop,
                         const float* __restrict__ Xproj,
                         float* __restrict__ out) {
    __shared__ unsigned short hbuf[2][16 * 256];       // double-buffered h (bf16)

    const int lane  = threadIdx.x & 31;
    const int wave  = threadIdx.x >> 5;                // 0..15 == N-tile of W_h
    const int b0    = blockIdx.x * 16;                 // batch rows owned
    const int row_l = lane & 15;
    const int hi    = (lane >> 4) & 1;
    const int kb    = hi ? 8 : 0;
    const int ncol  = wave * 16 + row_l;               // column this lane owns

    // h0 = pre_state rows [b0, b0+16)
    for (int i = threadIdx.x; i < 16 * 256; i += blockDim.x)
        hbuf[0][i] = f2bf_bits(pre_state[(size_t)(b0 + (i >> 8)) * 256 + (i & 255)]);

    // Resident W_h fragments (this wave's 16-col N-tile, all K) — 64 VGPRs.
    v16bf bW[KCH];
#pragma unroll
    for (int kc = 0; kc < KCH; ++kc)
        bW[kc] = load_bf16x16(Whp + ((size_t)((wave * KCH + kc) * 32 + lane) << 4));

    // Resident W_out fragments + bias (waves 0..3 only; wave-uniform branch).
    v16bf bO[KCH];
    float obias = 0.0f;
    if (wave < 4) {
#pragma unroll
        for (int kc = 0; kc < KCH; ++kc)
            bO[kc] = load_bf16x16(Wop + ((size_t)((wave * KCH + kc) * 32 + lane) << 4));
        obias = b_out[wave * 16 + row_l];
    }

    // Software pipeline: pre-load step 0's Xproj accumulator-init fragment.
    v8f cpre;
#pragma unroll
    for (int v = 0; v < 8; ++v)
        cpre[v] = Xproj[((size_t)0 * BATCH + b0 + v + hi * 8) * 256 + ncol];

    __syncthreads();

    for (int s = 0; s < SEQ_LEN; ++s) {
        const int cur = s & 1, nxt = cur ^ 1;

        v8f c = cpre;

        // Issue next step's Xproj loads now; waited on next iteration.
        if (s + 1 < SEQ_LEN) {
            const float* xn = Xproj + ((size_t)(s + 1) * BATCH + b0) * 256;
#pragma unroll
            for (int v = 0; v < 8; ++v)
                cpre[v] = xn[(size_t)(v + hi * 8) * 256 + ncol];
        }

        // Warm L2/L0 two steps ahead (global_prefetch_b8).
        if (s + 2 < SEQ_LEN) {
            __builtin_prefetch(Xproj + ((size_t)(s + 2) * BATCH + b0 + row_l) * 256
                                   + wave * 16, 0, 3);
        }

        // a_tile += h_{s-1} @ W_h^T   (K = 256)
#pragma unroll
        for (int kc = 0; kc < KCH; ++kc) {
            v16bf a = load_A_row(&hbuf[cur][row_l * 256 + kc * 32 + kb]);
            c = __builtin_amdgcn_wmma_f32_16x16x32_bf16(false, a, false, bW[kc],
                                                        (short)0, c, false, false);
        }

        // h_s = tanh(a); publish bf16 into the other LDS buffer
#pragma unroll
        for (int v = 0; v < 8; ++v)
            hbuf[nxt][(v + hi * 8) * 256 + ncol] = f2bf_bits(fast_tanh(c[v]));

        __syncthreads();   // all of h_s visible; old buffer free next iter

        // o_s = h_s @ W_out^T + b_out   (waves 0..3, one 16x16 tile each)
        if (wave < 4) {
            v8f oc;
#pragma unroll
            for (int v = 0; v < 8; ++v) oc[v] = obias;
#pragma unroll
            for (int kc = 0; kc < KCH; ++kc) {
                v16bf a = load_A_row(&hbuf[nxt][row_l * 256 + kc * 32 + kb]);
                oc = __builtin_amdgcn_wmma_f32_16x16x32_bf16(false, a, false, bO[kc],
                                                             (short)0, oc, false, false);
            }
#pragma unroll
            for (int v = 0; v < 8; ++v)
                out[(size_t)s * (BATCH * OUTPUT) + (b0 + v + hi * 8) * OUTPUT
                    + wave * 16 + row_l] = oc[v];
        }
    }
}

// ---- launcher -------------------------------------------------------------
extern "C" void kernel_launch(void* const* d_in, const int* in_sizes, int n_in,
                              void* d_out, int out_size, void* d_ws, size_t ws_size,
                              hipStream_t stream) {
    const float* x         = (const float*)d_in[0];
    const float* pre_state = (const float*)d_in[1];
    const float* W_in      = (const float*)d_in[2];
    const float* b_in      = (const float*)d_in[3];
    const float* W_out     = (const float*)d_in[4];
    const float* b_out     = (const float*)d_in[5];
    float* out             = (float*)d_out;

    // Workspace layout:
    //   [0, 288KB)        packed bf16 weights
    //   [1MB, 33MB)       xb  : x converted to bf16 (32 MB)
    //   [33MB, 97MB)      Xproj f32 (64 MB)
    unsigned short* Wxp = (unsigned short*)d_ws;
    unsigned short* Whp = Wxp + 256 * 256;
    unsigned short* Wop = Whp + 256 * 256;
    unsigned short* xb  = (unsigned short*)((char*)d_ws + (1ull << 20));
    float* Xproj        = (float*)((char*)d_ws + (1ull << 20) + (32ull << 20));

    // 0) prepack weights: 147456 elems, one thread each
    rnn_prepack<<<576, 256, 0, stream>>>(W_in, W_out, Wxp, Whp, Wop);

    // 0b) x -> bf16: 16.7M elems, 8 per thread
    rnn_x2bf<<<8192, 256, 0, stream>>>(x, xb);

    // 1) Xproj GEMM: 65536x256x256, 8 tiles/block -> 8192 blocks
    rnn_xproj<<<8192, 256, 0, stream>>>(xb, b_in, Wxp, Xproj);

    // 2) scan: 4 independent batch-slices, 16 waves each
    rnn_scan<<<4, 512, 0, stream>>>(pre_state, b_out, Whp, Wop, Xproj, out);
}
